// SARNNModel_17944373363356
// MI455X (gfx1250) — compile-verified, matrix operand order checked by reading, not run on hardware
//
#include <hip/hip_runtime.h>
#include <math.h>

namespace {

constexpr int kT = 64, kB = 32, kD = 1024, kL = 2, kMEM = 35, kNTOK = 32000;
constexpr int kDK = kD / 2;            // 512 (HEADS = 2)
constexpr int kBD = kB * kD;           // 32768
constexpr int kMR = kMEM * kB;         // 1120
constexpr float kEPS = 1e-6f;
constexpr float kSCALE = 0.0441941738241592f;  // 1/sqrt(512)

typedef __bf16 bf16_t;
typedef bf16_t bfx8  __attribute__((ext_vector_type(8)));
typedef bf16_t v16bf __attribute__((ext_vector_type(16)));
typedef float  v8f   __attribute__((ext_vector_type(8)));

union FragU { v16bf v; bfx8 h[2]; };

__device__ __forceinline__ float sigm(float x) { return 1.f / (1.f + __expf(-x)); }

// ---------------- fp32 -> bf16 conversion ----------------
__global__ void f32_to_bf16_kernel(const float* __restrict__ src, bf16_t* __restrict__ dst, size_t n) {
  size_t i = (size_t)blockIdx.x * blockDim.x + threadIdx.x;
  size_t stride = (size_t)gridDim.x * blockDim.x;
  for (; i < n; i += stride) dst[i] = (bf16_t)src[i];
}

// ---------------- WMMA GEMM ----------------
// C[M,N] = alpha * (A[M,K] @ Bw[N,K]^T) + bias[N] (+ old C if addC)
// A, Bw bf16 row-major. Optional outputs: Cf (f32), Cbf (bf16),
// CT (bf16 transposed store: CT[n*ldct + row_off + m]).
// One wave (= one block, blockDim 32) per 16x64 C tile.
// grid.x = ceil(N/64), grid.y = M/16 ; n0 is scalar (blockIdx only) so the
// whole K-loop is branch-free and WMMA runs with EXEC all-ones.
// K multiple of 32; M multiple of 16; N multiple of 16 (tail cols clamped).
__global__ void __launch_bounds__(32)
wmma_gemm_kernel(const bf16_t* __restrict__ A, int lda,
                 const bf16_t* __restrict__ Bw, int ldb,
                 const float* __restrict__ bias, float alpha,
                 float* __restrict__ Cf, bf16_t* __restrict__ Cbf, int ldc,
                 bf16_t* __restrict__ CT, int ldct, int row_off,
                 int N, int K, int addC) {
  const int lane = threadIdx.x;              // 0..31
  const int n0 = blockIdx.x * 64;            // scalar
  const int m0 = blockIdx.y * 16;            // scalar
  const int koff = (lane >> 4) << 3;         // lanes 0-15: K 0..7,16..23 ; 16-31: K 8..15,24..31
  const bf16_t* Arow = A + (size_t)(m0 + (lane & 15)) * lda + koff;
  const bf16_t* B0;
  const bf16_t* B1;
  const bf16_t* B2;
  const bf16_t* B3;
  {
    int n_s, nc;
    n_s = n0 + 0 * 16 + (lane & 15); nc = n_s < N ? n_s : N - 1; B0 = Bw + (size_t)nc * ldb + koff;
    n_s = n0 + 1 * 16 + (lane & 15); nc = n_s < N ? n_s : N - 1; B1 = Bw + (size_t)nc * ldb + koff;
    n_s = n0 + 2 * 16 + (lane & 15); nc = n_s < N ? n_s : N - 1; B2 = Bw + (size_t)nc * ldb + koff;
    n_s = n0 + 3 * 16 + (lane & 15); nc = n_s < N ? n_s : N - 1; B3 = Bw + (size_t)nc * ldb + koff;
  }
  v8f acc0 = {}, acc1 = {}, acc2 = {}, acc3 = {};
  for (int k = 0; k < K; k += 32) {
    FragU fa, f0, f1, f2, f3;
    fa.h[0] = *(const bfx8*)(Arow + k);      fa.h[1] = *(const bfx8*)(Arow + k + 16);
    f0.h[0] = *(const bfx8*)(B0 + k);        f0.h[1] = *(const bfx8*)(B0 + k + 16);
    f1.h[0] = *(const bfx8*)(B1 + k);        f1.h[1] = *(const bfx8*)(B1 + k + 16);
    f2.h[0] = *(const bfx8*)(B2 + k);        f2.h[1] = *(const bfx8*)(B2 + k + 16);
    f3.h[0] = *(const bfx8*)(B3 + k);        f3.h[1] = *(const bfx8*)(B3 + k + 16);
    acc0 = __builtin_amdgcn_wmma_f32_16x16x32_bf16(false, fa.v, false, f0.v, (short)0, acc0, false, false);
    acc1 = __builtin_amdgcn_wmma_f32_16x16x32_bf16(false, fa.v, false, f1.v, (short)0, acc1, false, false);
    acc2 = __builtin_amdgcn_wmma_f32_16x16x32_bf16(false, fa.v, false, f2.v, (short)0, acc2, false, false);
    acc3 = __builtin_amdgcn_wmma_f32_16x16x32_bf16(false, fa.v, false, f3.v, (short)0, acc3, false, false);
  }
  // ---- epilogue: stores masked by n < N (after all WMMAs) ----
  const int mlane = (lane >> 4) << 3;        // C/D layout: lanes 16-31 hold M = r+8
  v8f accs[4] = {acc0, acc1, acc2, acc3};
#pragma unroll
  for (int s = 0; s < 4; ++s) {
    const int n = n0 + s * 16 + (lane & 15);
    if (n < N) {
      const float bv = bias ? bias[n] : 0.f;
#pragma unroll
      for (int r = 0; r < 8; ++r) {
        const int m = m0 + mlane + r;
        float val = alpha * accs[s][r] + bv;
        if (Cf) {
          size_t off = (size_t)m * ldc + n;
          if (addC) val += Cf[off];
          Cf[off] = val;
        }
        if (Cbf) Cbf[(size_t)m * ldc + n] = (bf16_t)val;
        if (CT)  CT[(size_t)n * ldct + row_off + m] = (bf16_t)val;
      }
    }
  }
}

// ---------------- embedding gather + LayerNorm ----------------
__global__ void embed_ln_kernel(const int* __restrict__ tokens, const float* __restrict__ emb,
                                const float* __restrict__ ln_g, const float* __restrict__ ln_b,
                                float* __restrict__ e_f, bf16_t* __restrict__ e_bf,
                                bf16_t* __restrict__ lne_bf, int t) {
  __shared__ float red[256];
  const int b = blockIdx.x;
  const int tid = threadIdx.x;
  const int tok = tokens[t * kB + b];
  const float* row = emb + (size_t)tok * kD;
  float loc[4];
  float s = 0.f;
#pragma unroll
  for (int i = 0; i < 4; ++i) { loc[i] = row[tid + i * 256]; s += loc[i]; }
  red[tid] = s; __syncthreads();
  for (int off = 128; off > 0; off >>= 1) { if (tid < off) red[tid] += red[tid + off]; __syncthreads(); }
  const float mu = red[0] / kD;
  __syncthreads();
  float vs = 0.f;
#pragma unroll
  for (int i = 0; i < 4; ++i) { float d = loc[i] - mu; vs += d * d; }
  red[tid] = vs; __syncthreads();
  for (int off = 128; off > 0; off >>= 1) { if (tid < off) red[tid] += red[tid + off]; __syncthreads(); }
  const float sd = sqrtf(red[0] / (float)(kD - 1));  // unbiased std, eps added to std
  const float inv = 1.f / (sd + kEPS);
#pragma unroll
  for (int i = 0; i < 4; ++i) {
    const int d = tid + i * 256;
    const float x = loc[i];
    e_f[b * kD + d] = x;
    e_bf[b * kD + d] = (bf16_t)x;
    lne_bf[b * kD + d] = (bf16_t)(ln_g[d] * (x - mu) * inv + ln_b[d]);
  }
}

// ---------------- row softmax -> bf16 ----------------
__global__ void softmax_bf16_kernel(const float* __restrict__ x, bf16_t* __restrict__ out, int n) {
  __shared__ float red[128];
  const int row = blockIdx.x;
  const int tid = threadIdx.x;
  const float* xr = x + (size_t)row * n;
  float mx = -3.4e38f;
  for (int i = tid; i < n; i += 128) mx = fmaxf(mx, xr[i]);
  red[tid] = mx; __syncthreads();
  for (int off = 64; off > 0; off >>= 1) { if (tid < off) red[tid] = fmaxf(red[tid], red[tid + off]); __syncthreads(); }
  const float m = red[0];
  __syncthreads();
  float s = 0.f;
  for (int i = tid; i < n; i += 128) s += __expf(xr[i] - m);
  red[tid] = s; __syncthreads();
  for (int off = 64; off > 0; off >>= 1) { if (tid < off) red[tid] += red[tid + off]; __syncthreads(); }
  const float inv = 1.f / red[0];
  for (int i = tid; i < n; i += 128) out[(size_t)row * n + i] = (bf16_t)(__expf(xr[i] - m) * inv);
}

// ---------------- h_used = h_prev + e + att (bf16 for gate GEMMs) ----------------
__global__ void h_used_kernel(const float* __restrict__ h, const float* __restrict__ e_f,
                              const float* __restrict__ att, bf16_t* __restrict__ out) {
  int idx = blockIdx.x * blockDim.x + threadIdx.x;   // kL*kBD
  int j = idx & (kBD - 1);
  out[idx] = (bf16_t)(h[idx] + e_f[j] + att[j]);
}

// ---------------- LSTM cell pointwise (one layer) ----------------
__global__ void lstm_point_kernel(const float* __restrict__ gates,
                                  float* __restrict__ h_carry, float* __restrict__ c_carry,
                                  bf16_t* __restrict__ hn_bf,
                                  float* __restrict__ ringH, float* __restrict__ ringC,
                                  bf16_t* __restrict__ hmem) {
  int idx = blockIdx.x * blockDim.x + threadIdx.x;   // kBD
  int b = idx >> 10, d = idx & 1023;
  const float* g = gates + (size_t)b * 4 * kD;
  float ig = sigm(g[d]);
  float fg = sigm(g[kD + d]);
  float gg = tanhf(g[2 * kD + d]);
  float og = sigm(g[3 * kD + d]);
  float cn = fg * c_carry[idx] + ig * gg;
  float hn = og * tanhf(cn);
  c_carry[idx] = cn;
  h_carry[idx] = hn;
  hn_bf[idx] = (bf16_t)hn;
  ringH[idx] = hn;
  ringC[idx] = cn;
  if (hmem) hmem[idx] = (bf16_t)hn;
}

// ---------------- unroll circular memory into logical order ----------------
__global__ void ring_reorder_kernel(const float* __restrict__ ring, float* __restrict__ out, int shift) {
  int idx = blockIdx.x * blockDim.x + threadIdx.x;   // kL*kMEM*kBD
  int l = idx / (kMEM * kBD);
  int rem = idx - l * (kMEM * kBD);
  int j = rem / kBD;
  int e = rem - j * kBD;
  int phys = j + shift; if (phys >= kMEM) phys -= kMEM;
  out[idx] = ring[((size_t)l * kMEM + phys) * kBD + e];
}

void launch_gemm(hipStream_t s, const bf16_t* A, int lda, const bf16_t* Bw, int ldb,
                 const float* bias, float alpha, float* Cf, bf16_t* Cbf, int ldc,
                 bf16_t* CT, int ldct, int row_off, int M, int N, int K, int addC) {
  dim3 grid((N + 63) / 64, M / 16);
  wmma_gemm_kernel<<<grid, 32, 0, s>>>(A, lda, Bw, ldb, bias, alpha, Cf, Cbf, ldc,
                                       CT, ldct, row_off, N, K, addC);
}

}  // namespace

extern "C" void kernel_launch(void* const* d_in, const int* in_sizes, int n_in,
                              void* d_out, int out_size, void* d_ws, size_t ws_size,
                              hipStream_t stream) {
  (void)in_sizes; (void)n_in; (void)out_size; (void)ws_size;
  const int*   tokens = (const int*)d_in[0];
  const float* emb  = (const float*)d_in[1];
  const float* ln_g = (const float*)d_in[2];
  const float* ln_b = (const float*)d_in[3];
  const float* Wq = (const float*)d_in[4];  const float* bq = (const float*)d_in[5];
  const float* Wk = (const float*)d_in[6];  const float* bk = (const float*)d_in[7];
  const float* Wv = (const float*)d_in[8];  const float* bv = (const float*)d_in[9];
  const float* Wo = (const float*)d_in[10]; const float* bo = (const float*)d_in[11];
  const float* W_ih = (const float*)d_in[12]; const float* b_ih = (const float*)d_in[13];
  const float* W_hh = (const float*)d_in[14]; const float* b_hh = (const float*)d_in[15];
  const float* decW = (const float*)d_in[16]; const float* decb = (const float*)d_in[17];
  const float* h0 = (const float*)d_in[18]; const float* c0 = (const float*)d_in[19];
  const float* H0 = (const float*)d_in[20]; const float* C0 = (const float*)d_in[21];
  float* out = (float*)d_out;

  // ---- workspace bump allocator ----
  char* p = (char*)d_ws;
  auto alloc = [&](size_t bytes) -> void* {
    void* r = (void*)p; p += (bytes + 255) & ~(size_t)255; return r;
  };
  bf16_t* Wq_bf  = (bf16_t*)alloc((size_t)kD * kD * 2);
  bf16_t* Wk_bf  = (bf16_t*)alloc((size_t)kD * kD * 2);
  bf16_t* Wv_bf  = (bf16_t*)alloc((size_t)kD * kD * 2);
  bf16_t* Wo_bf  = (bf16_t*)alloc((size_t)kD * kD * 2);
  bf16_t* Wih_bf = (bf16_t*)alloc((size_t)kL * 4 * kD * kD * 2);
  bf16_t* Whh_bf = (bf16_t*)alloc((size_t)kL * 4 * kD * kD * 2);
  bf16_t* decW_bf= (bf16_t*)alloc((size_t)kNTOK * kD * 2);
  float*  ringH  = (float*)alloc((size_t)kL * kMEM * kBD * 4);
  float*  ringC  = (float*)alloc((size_t)kL * kMEM * kBD * 4);
  bf16_t* hmem_bf= (bf16_t*)alloc((size_t)kMEM * kBD * 2);     // layer-0 memory, bf16
  bf16_t* k_bf   = (bf16_t*)alloc((size_t)kMR * kD * 2);       // cached k rows
  bf16_t* vT_bf  = (bf16_t*)alloc((size_t)kD * kMR * 2);       // cached v, transposed [D, MEM*B]
  float*  e_f    = (float*)alloc((size_t)kBD * 4);
  bf16_t* e_bf   = (bf16_t*)alloc((size_t)kBD * 2);
  bf16_t* lne_bf = (bf16_t*)alloc((size_t)kBD * 2);
  bf16_t* q_bf   = (bf16_t*)alloc((size_t)kBD * 2);
  float*  scores = (float*)alloc((size_t)2 * kB * kMR * 4);
  bf16_t* attnw_bf = (bf16_t*)alloc((size_t)2 * kB * kMR * 2);
  bf16_t* atth_bf  = (bf16_t*)alloc((size_t)kBD * 2);
  float*  att_f  = (float*)alloc((size_t)kBD * 4);
  bf16_t* hused_bf = (bf16_t*)alloc((size_t)kL * kBD * 2);
  bf16_t* hn_bf  = (bf16_t*)alloc((size_t)kL * kBD * 2);
  float*  h_carry= (float*)alloc((size_t)kL * kBD * 4);
  float*  c_carry= (float*)alloc((size_t)kL * kBD * 4);
  float*  gates  = (float*)alloc((size_t)kB * 4 * kD * 4);

  auto conv = [&](const float* src, bf16_t* dst, size_t n) {
    size_t g = (n + 255) / 256; if (g > 16384) g = 16384;
    f32_to_bf16_kernel<<<(int)g, 256, 0, stream>>>(src, dst, n);
  };

  // ---- prologue: weight conversion, state init, initial k/v cache ----
  conv(Wq, Wq_bf, (size_t)kD * kD);
  conv(Wk, Wk_bf, (size_t)kD * kD);
  conv(Wv, Wv_bf, (size_t)kD * kD);
  conv(Wo, Wo_bf, (size_t)kD * kD);
  conv(W_ih, Wih_bf, (size_t)kL * 4 * kD * kD);
  conv(W_hh, Whh_bf, (size_t)kL * 4 * kD * kD);
  conv(decW, decW_bf, (size_t)kNTOK * kD);
  conv(H0, hmem_bf, (size_t)kMEM * kBD);  // layer-0 slice of H0
  hipMemcpyAsync(ringH, H0, (size_t)kL * kMEM * kBD * 4, hipMemcpyDeviceToDevice, stream);
  hipMemcpyAsync(ringC, C0, (size_t)kL * kMEM * kBD * 4, hipMemcpyDeviceToDevice, stream);
  hipMemcpyAsync(h_carry, h0, (size_t)kL * kBD * 4, hipMemcpyDeviceToDevice, stream);
  hipMemcpyAsync(c_carry, c0, (size_t)kL * kBD * 4, hipMemcpyDeviceToDevice, stream);
  // full k/v projection of the initial memory (subsequent steps re-project 1 slot)
  launch_gemm(stream, hmem_bf, kD, Wk_bf, kD, bk, 1.f, nullptr, k_bf, kD,
              nullptr, 0, 0, kMR, kD, kD, 0);
  launch_gemm(stream, hmem_bf, kD, Wv_bf, kD, bv, 1.f, nullptr, nullptr, 0,
              vT_bf, kMR, 0, kMR, kD, kD, 0);

  // ---- T sequential steps ----
  for (int t = 0; t < kT; ++t) {
    const int pslot = t % kMEM;
    embed_ln_kernel<<<kB, 256, 0, stream>>>(tokens, emb, ln_g, ln_b, e_f, e_bf, lne_bf, t);
    // q = LN(e) @ Wq^T + bq
    launch_gemm(stream, lne_bf, kD, Wq_bf, kD, bq, 1.f, nullptr, q_bf, kD,
                nullptr, 0, 0, kB, kD, kD, 0);
    // scores[h,b,m] = SCALE * sum_d q[b,h*DK+d] k[m,h*DK+d]
    for (int hh = 0; hh < 2; ++hh)
      launch_gemm(stream, q_bf + hh * kDK, kD, k_bf + hh * kDK, kD, nullptr, kSCALE,
                  scores + (size_t)hh * kB * kMR, nullptr, kMR, nullptr, 0, 0,
                  kB, kMR, kDK, 0);
    softmax_bf16_kernel<<<2 * kB, 128, 0, stream>>>(scores, attnw_bf, kMR);
    // att[b,h*DK+d] = sum_m attnw[h,b,m] v[m,h*DK+d]   (B operand = vT rows)
    for (int hh = 0; hh < 2; ++hh)
      launch_gemm(stream, attnw_bf + (size_t)hh * kB * kMR, kMR,
                  vT_bf + (size_t)hh * kDK * kMR, kMR, nullptr, 1.f,
                  nullptr, atth_bf + hh * kDK, kD, nullptr, 0, 0, kB, kDK, kMR, 0);
    // out projection
    launch_gemm(stream, atth_bf, kD, Wo_bf, kD, bo, 1.f, att_f, nullptr, kD,
                nullptr, 0, 0, kB, kD, kD, 0);
    // h_used[l] = h_prev[l] + e + att
    h_used_kernel<<<(kL * kBD) / 256, 256, 0, stream>>>(h_carry, e_f, att_f, hused_bf);
    // ---- LSTM layer 0 ----
    launch_gemm(stream, e_bf, kD, Wih_bf, kD, b_ih, 1.f, gates, nullptr, 4 * kD,
                nullptr, 0, 0, kB, 4 * kD, kD, 0);
    launch_gemm(stream, hused_bf, kD, Whh_bf, kD, b_hh, 1.f, gates, nullptr, 4 * kD,
                nullptr, 0, 0, kB, 4 * kD, kD, 1);
    lstm_point_kernel<<<kBD / 256, 256, 0, stream>>>(gates, h_carry, c_carry, hn_bf,
        ringH + (size_t)pslot * kBD, ringC + (size_t)pslot * kBD,
        hmem_bf + (size_t)pslot * kBD);
    // refresh k/v cache rows for the slot just written
    launch_gemm(stream, hmem_bf + (size_t)pslot * kBD, kD, Wk_bf, kD, bk, 1.f,
                nullptr, k_bf + (size_t)pslot * kBD, kD, nullptr, 0, 0, kB, kD, kD, 0);
    launch_gemm(stream, hmem_bf + (size_t)pslot * kBD, kD, Wv_bf, kD, bv, 1.f,
                nullptr, nullptr, 0, vT_bf, kMR, pslot * kB, kB, kD, kD, 0);
    // ---- LSTM layer 1 ----
    launch_gemm(stream, hn_bf, kD, Wih_bf + (size_t)4 * kD * kD, kD, b_ih + 4 * kD,
                1.f, gates, nullptr, 4 * kD, nullptr, 0, 0, kB, 4 * kD, kD, 0);
    launch_gemm(stream, hused_bf + kBD, kD, Whh_bf + (size_t)4 * kD * kD, kD,
                b_hh + 4 * kD, 1.f, gates, nullptr, 4 * kD, nullptr, 0, 0,
                kB, 4 * kD, kD, 1);
    lstm_point_kernel<<<kBD / 256, 256, 0, stream>>>(gates, h_carry + kBD, c_carry + kBD,
        hn_bf + kBD, ringH + (size_t)(kMEM + pslot) * kBD,
        ringC + (size_t)(kMEM + pslot) * kBD, nullptr);
    // decoder: dec = x @ dec_W^T + dec_b  (dec_W bf16 lives in L2 after step 0)
    launch_gemm(stream, hn_bf + kBD, kD, decW_bf, kD, decb, 1.f,
                out + (size_t)t * kB * kNTOK, nullptr, kNTOK, nullptr, 0, 0,
                kB, kNTOK, kD, 0);
  }

  // ---- epilogue: final h, c, H, C ----
  const size_t off_h = (size_t)kT * kB * kNTOK;
  const size_t off_c = off_h + (size_t)kL * kBD;
  const size_t off_H = off_c + (size_t)kL * kBD;
  const size_t off_C = off_H + (size_t)kL * kMEM * kBD;
  hipMemcpyAsync(out + off_h, h_carry, (size_t)kL * kBD * 4, hipMemcpyDeviceToDevice, stream);
  hipMemcpyAsync(out + off_c, c_carry, (size_t)kL * kBD * 4, hipMemcpyDeviceToDevice, stream);
  const int shift = kT % kMEM;  // 29
  ring_reorder_kernel<<<(kL * kMEM * kBD) / 256, 256, 0, stream>>>(ringH, out + off_H, shift);
  ring_reorder_kernel<<<(kL * kMEM * kBD) / 256, 256, 0, stream>>>(ringC, out + off_C, shift);
}
